// GatedMultiScaleRetention_8675833938651
// MI455X (gfx1250) — compile-verified, matrix operand order checked by reading, not run on hardware
//
#include <hip/hip_runtime.h>

#define D_MODEL 2048
#define N_HEADS 16
#define D_HEAD 128
#define SQLEN 2048
#define QKV_N (3 * D_MODEL)

typedef __attribute__((ext_vector_type(16))) __bf16 v16bf;
typedef __attribute__((ext_vector_type(8)))  __bf16 v8bf;
typedef __attribute__((ext_vector_type(8)))  float  v8f;

// ---------------------------------------------------------------------------
// WMMA helpers (NT layout).
// 16-bit A-matrix 16x32 (and B^T, which matches when B is stored N x K):
//   lane l (0..15):  row M = l,    K chunks {0..7}  and {16..23}
//   lane l (16..31): row M = l-16, K chunks {8..15} and {24..31}
// => two contiguous 16-byte loads per lane.
// ---------------------------------------------------------------------------
__device__ __forceinline__ v16bf load_frag_nt(const __bf16* base, int ld, int lane) {
  const int r  = lane & 15;
  const int kb = (lane >> 4) << 3;  // 0 or 8
  const __bf16* p = base + (size_t)r * ld + kb;
  v8bf lo = *reinterpret_cast<const v8bf*>(p);
  v8bf hi = *reinterpret_cast<const v8bf*>(p + 16);
  v16bf f;
#pragma unroll
  for (int i = 0; i < 8; ++i) { f[i] = lo[i]; f[i + 8] = hi[i]; }
  return f;
}

__device__ __forceinline__ v8f wmma_bf16(v16bf a, v16bf b, v8f c) {
  return __builtin_amdgcn_wmma_f32_16x16x32_bf16(false, a, false, b, (short)0, c,
                                                 false, false);
}

// CDNA5 async memory->LDS DMA (ASYNCcnt-tracked); bypasses VGPRs entirely.
__device__ __forceinline__ void async_copy_b128(unsigned lds_off,
                                                const __bf16* gptr) {
  asm volatile("global_load_async_to_lds_b128 %0, %1, off"
               :
               : "v"(lds_off), "v"(gptr)
               : "memory");
}

// ---------------------------------------------------------------------------
// Generic NT GEMM: C[M,N] (f32) = A[M,K] (bf16) * B[N,K]^T (bf16)
// block = 256 threads (8 waves, 2x4), block tile 64(M) x 128(N), K step 32.
// Double-buffered LDS filled with global_load_async_to_lds_b128.
// ---------------------------------------------------------------------------
__global__ __launch_bounds__(256) void gemm_nt_wmma(
    float* __restrict__ C, const __bf16* __restrict__ A,
    const __bf16* __restrict__ B, int M, int N, int K) {
  __shared__ __attribute__((aligned(16))) __bf16 As[2][64][32];
  __shared__ __attribute__((aligned(16))) __bf16 Bs[2][128][32];
  const int tid  = threadIdx.x;
  const int lane = tid & 31;
  const int w    = tid >> 5;
  const int bm   = blockIdx.y * 64;
  const int bn   = blockIdx.x * 128;
  const int m0   = (w >> 2) * 32;
  const int n0   = (w & 3) * 32;

  // per-thread copy slots: A = 1 x b128, B = 2 x b128
  const int ar = tid >> 2, ac = (tid & 3) << 3;   // A rows 0..63
  const int br = tid >> 2, bc = (tid & 3) << 3;   // B rows 0..63 and 64..127
  const __bf16* gA  = A + (size_t)(bm + ar) * K + ac;
  const __bf16* gB0 = B + (size_t)(bn + br) * K + bc;
  const __bf16* gB1 = B + (size_t)(bn + br + 64) * K + bc;

  v8f acc[2][2] = {};
  const int nk = K >> 5;

  // prologue: panel 0 into buffer 0
  async_copy_b128((unsigned)(size_t)&As[0][ar][ac], gA);
  async_copy_b128((unsigned)(size_t)&Bs[0][br][bc], gB0);
  async_copy_b128((unsigned)(size_t)&Bs[0][br + 64][bc], gB1);

  for (int ks = 0; ks < nk; ++ks) {
    const int buf = ks & 1;
    if (ks + 1 < nk) {
      const int koff = (ks + 1) << 5;
      async_copy_b128((unsigned)(size_t)&As[buf ^ 1][ar][ac], gA + koff);
      async_copy_b128((unsigned)(size_t)&Bs[buf ^ 1][br][bc], gB0 + koff);
      async_copy_b128((unsigned)(size_t)&Bs[buf ^ 1][br + 64][bc], gB1 + koff);
      // 3 newest (next panel) may stay in flight; current panel has landed.
      asm volatile("s_wait_asynccnt 0x3" ::: "memory");
    } else {
      asm volatile("s_wait_asynccnt 0x0" ::: "memory");
    }
    __syncthreads();  // all waves' contributions to this buffer are visible

    v16bf a0 = load_frag_nt(&As[buf][m0][0],      32, lane);
    v16bf a1 = load_frag_nt(&As[buf][m0 + 16][0], 32, lane);
    v16bf b0 = load_frag_nt(&Bs[buf][n0][0],      32, lane);
    v16bf b1 = load_frag_nt(&Bs[buf][n0 + 16][0], 32, lane);
    acc[0][0] = wmma_bf16(a0, b0, acc[0][0]);
    acc[0][1] = wmma_bf16(a0, b1, acc[0][1]);
    acc[1][0] = wmma_bf16(a1, b0, acc[1][0]);
    acc[1][1] = wmma_bf16(a1, b1, acc[1][1]);

    __syncthreads();  // protect buffer from next-next async overwrite
  }

  const int hi8 = (lane >> 4) << 3;
  const int cn  = lane & 15;
#pragma unroll
  for (int mm = 0; mm < 2; ++mm)
#pragma unroll
    for (int nn = 0; nn < 2; ++nn)
#pragma unroll
      for (int r = 0; r < 8; ++r)
        C[(size_t)(bm + m0 + mm * 16 + r + hi8) * N + bn + n0 + nn * 16 + cn] =
            acc[mm][nn][r];
}

// ---------------------------------------------------------------------------
// Retention attention: one wave per (head, 16-row i-block).
//  S = (Q K^T) .* Delta  -> LDS relayout -> ret += S V^T(rows=e) ; plus state term
// ---------------------------------------------------------------------------
__global__ __launch_bounds__(256) void retention_attn_wmma(
    float* __restrict__ ret, const __bf16* __restrict__ qg,
    const __bf16* __restrict__ kg, const __bf16* __restrict__ vtg,
    const __bf16* __restrict__ stg, const float* __restrict__ ca) {
  __shared__ __attribute__((aligned(16))) __bf16 Sbuf[8][16 * 32];
  const int lane = threadIdx.x & 31;
  const int w    = threadIdx.x >> 5;
  const int ib   = blockIdx.x * 8 + w;   // 0..127
  const int h    = blockIdx.y;
  const __bf16* q   = qg  + (size_t)h * SQLEN * D_HEAD;
  const __bf16* k   = kg  + (size_t)h * SQLEN * D_HEAD;
  const __bf16* vt  = vtg + (size_t)h * D_HEAD * SQLEN;   // [e][t]
  const __bf16* st  = stg + (size_t)h * D_HEAD * D_HEAD;  // [e][d]
  const float*  cah = ca  + (size_t)h * SQLEN;
  const int hi8 = (lane >> 4) << 3;
  const int cn  = lane & 15;

  v16bf qf[4];
#pragma unroll
  for (int kk = 0; kk < 4; ++kk)
    qf[kk] = load_frag_nt(q + (size_t)(ib * 16) * D_HEAD + kk * 32, D_HEAD, lane);

  float cai[8];
#pragma unroll
  for (int r = 0; r < 8; ++r) cai[r] = cah[ib * 16 + r + hi8];

  v8f acc[8] = {};

  const int jmax = ib >> 1;
  for (int jb2 = 0; jb2 <= jmax; ++jb2) {
#pragma unroll
    for (int hf = 0; hf < 2; ++hf) {
      const int j0 = jb2 * 32 + hf * 16;
      v8f s = {};
#pragma unroll
      for (int kk = 0; kk < 4; ++kk) {
        v16bf kf = load_frag_nt(k + (size_t)j0 * D_HEAD + kk * 32, D_HEAD, lane);
        s = wmma_bf16(qf[kk], kf, s);
      }
      const int   j   = j0 + cn;
      const float caj = cah[j];
#pragma unroll
      for (int r = 0; r < 8; ++r) {
        const int   i = ib * 16 + r + hi8;
        const float d = (j <= i) ? __expf(cai[r] - caj) : 0.0f;
        Sbuf[w][(r + hi8) * 32 + hf * 16 + cn] = (__bf16)(s[r] * d);
      }
    }
    asm volatile("s_wait_dscnt 0" ::: "memory");  // LDS relayout: D-frag -> A-frag
    v16bf sf = load_frag_nt(&Sbuf[w][0], 32, lane);
#pragma unroll
    for (int e = 0; e < 8; ++e) {
      v16bf vf = load_frag_nt(vt + (size_t)(e * 16) * SQLEN + jb2 * 32, SQLEN, lane);
      acc[e] = wmma_bf16(sf, vf, acc[e]);
    }
  }

  // carry-in from previous state: ret += diag(exp(ca_i)) * (Q @ state)
#pragma unroll
  for (int e = 0; e < 8; ++e) {
    v8f t = {};
#pragma unroll
    for (int kk = 0; kk < 4; ++kk) {
      v16bf sb = load_frag_nt(st + (size_t)(e * 16) * D_HEAD + kk * 32, D_HEAD, lane);
      t = wmma_bf16(qf[kk], sb, t);
    }
#pragma unroll
    for (int r = 0; r < 8; ++r) acc[e][r] += __expf(cai[r]) * t[r];
  }

#pragma unroll
  for (int e = 0; e < 8; ++e)
#pragma unroll
    for (int r = 0; r < 8; ++r)
      ret[(size_t)(ib * 16 + r + hi8) * D_MODEL + h * D_HEAD + e * 16 + cn] =
          acc[e][r];
}

// ---------------------------------------------------------------------------
// alphas: a_raw[h,t] = alpha_base[h] * 8 * sigmoid(x[t] . W_alpha[h] + b[h])
// ---------------------------------------------------------------------------
__global__ __launch_bounds__(256) void alpha_raw_kernel(
    float* __restrict__ a_raw, const float* __restrict__ x,
    const float* __restrict__ Wa, const float* __restrict__ ba,
    const float* __restrict__ abase) {
  const int lane = threadIdx.x & 31;
  const int w    = threadIdx.x >> 5;
  const int idx  = blockIdx.x * 8 + w;   // h*SQLEN + t
  const int h    = idx >> 11;
  float s = 0.f;
  const float* xr = x  + (size_t)(idx & (SQLEN - 1)) * D_MODEL;
  const float* wr = Wa + (size_t)h * D_MODEL;
  for (int j = lane; j < D_MODEL; j += 32) s += xr[j] * wr[j];
#pragma unroll
  for (int off = 16; off; off >>= 1) s += __shfl_xor(s, off, 32);
  if (lane == 0) {
    const float sg = 1.0f / (1.0f + __expf(-(s + ba[h])));
    a_raw[idx] = abase[h] * 8.0f * sg;
  }
}

__global__ void cumsum_kernel(float* __restrict__ ca, const float* __restrict__ a_raw) {
  if (threadIdx.x != 0) return;
  const int h = blockIdx.x;
  float run = 0.f;
  for (int t = 0; t < SQLEN; ++t) {
    run += a_raw[h * SQLEN + t];
    ca[h * SQLEN + t] = run;
  }
}

// ---------------------------------------------------------------------------
// xpos rotate q (fwd) / k (inv, * (1-exp(alpha))), write bf16 q/k [h][t][d]
// and transposed v [h][e][t].
// ---------------------------------------------------------------------------
__global__ __launch_bounds__(256) void qkv_prep_kernel(
    __bf16* __restrict__ qb, __bf16* __restrict__ kb, __bf16* __restrict__ vtb,
    const float* __restrict__ qkv, const float* __restrict__ a_raw,
    const int* __restrict__ offp) {
  const int idx = blockIdx.x * 256 + threadIdx.x;  // h*SQLEN*64 + t*64 + i
  const int i = idx & 63;
  const int t = (idx >> 6) & (SQLEN - 1);
  const int h = idx >> 17;
  const float pos  = (float)t + (float)offp[0];
  const float freq = __expf(-((float)i / 64.0f) * 9.210340371976184f);  // 10000^-i/64
  float sn, cs;
  __sincosf(pos * freq, &sn, &cs);
  const float zeta = ((float)(2 * i) + 0.4f * 128.0f) / (1.4f * 128.0f);
  const float sc   = __expf(__logf(zeta) * (pos * (1.0f / 512.0f)));
  const size_t qoff = ((size_t)(t * 3 + 0) * N_HEADS + h) * D_HEAD;
  const size_t koff = ((size_t)(t * 3 + 1) * N_HEADS + h) * D_HEAD;
  const size_t voff = ((size_t)(t * 3 + 2) * N_HEADS + h) * D_HEAD;
  const float q1 = qkv[qoff + i], q2 = qkv[qoff + i + 64];
  const float k1 = qkv[koff + i], k2 = qkv[koff + i + 64];
  const float v1 = qkv[voff + i], v2 = qkv[voff + i + 64];
  const float fk = (1.0f - __expf(a_raw[h * SQLEN + t])) / sc;  // inv-scale * decay
  __bf16* qd = qb + ((size_t)h * SQLEN + t) * D_HEAD;
  __bf16* kd = kb + ((size_t)h * SQLEN + t) * D_HEAD;
  qd[i]      = (__bf16)((q1 * cs - q2 * sn) * sc);
  qd[i + 64] = (__bf16)((q2 * cs + q1 * sn) * sc);
  kd[i]      = (__bf16)((k1 * cs - k2 * sn) * fk);
  kd[i + 64] = (__bf16)((k2 * cs + k1 * sn) * fk);
  __bf16* vd = vtb + (size_t)h * D_HEAD * SQLEN;
  vd[(size_t)i * SQLEN + t]        = (__bf16)v1;
  vd[(size_t)(i + 64) * SQLEN + t] = (__bf16)v2;
}

// ---------------------------------------------------------------------------
// new_state[h,d,e] = exp(ca[h,last]) * state + sum_t k[h,t,d]*exp(ca_last-ca_t)*v[h,t,e]
// ---------------------------------------------------------------------------
__global__ __launch_bounds__(128) void newstate_kernel(
    float* __restrict__ outS, const __bf16* __restrict__ kb,
    const float* __restrict__ qkv, const float* __restrict__ ca,
    const float* __restrict__ state) {
  const int h = blockIdx.y;
  const int d = blockIdx.x;
  const int e = threadIdx.x;
  const float calast = ca[h * SQLEN + SQLEN - 1];
  float acc = 0.f;
  for (int t = 0; t < SQLEN; ++t) {
    if ((t & 7) == 0)
      __builtin_prefetch(qkv + ((size_t)((t + 8) * 3 + 2) * N_HEADS + h) * D_HEAD + e,
                         0, 1);
    const float kv  = (float)kb[((size_t)h * SQLEN + t) * D_HEAD + d];
    const float wgt = __expf(calast - ca[h * SQLEN + t]);
    const float vv  = qkv[((size_t)(t * 3 + 2) * N_HEADS + h) * D_HEAD + e];
    acc += kv * wgt * vv;
  }
  const size_t o = ((size_t)h * D_HEAD + d) * D_HEAD + e;
  outS[o] = __expf(calast) * state[o] + acc;
}

// ---------------------------------------------------------------------------
// group norm over D_HEAD per (t, head); writes fp32 + bf16 copies
// ---------------------------------------------------------------------------
__global__ __launch_bounds__(256) void groupnorm_kernel(
    float* __restrict__ rg, __bf16* __restrict__ rgb,
    const float* __restrict__ ret, const float* __restrict__ wgt,
    const float* __restrict__ bia) {
  const int lane = threadIdx.x & 31;
  const int w    = threadIdx.x >> 5;
  const int idx  = blockIdx.x * 8 + w;  // t*16 + h
  const int t = idx >> 4;
  const int h = idx & 15;
  const size_t base = (size_t)t * D_MODEL + h * D_HEAD;
  float v[4], sum = 0.f;
#pragma unroll
  for (int c = 0; c < 4; ++c) { v[c] = ret[base + c * 32 + lane]; sum += v[c]; }
#pragma unroll
  for (int off = 16; off; off >>= 1) sum += __shfl_xor(sum, off, 32);
  const float mu = sum * (1.0f / 128.0f);
  float sq = 0.f;
#pragma unroll
  for (int c = 0; c < 4; ++c) { const float d = v[c] - mu; sq += d * d; }
#pragma unroll
  for (int off = 16; off; off >>= 1) sq += __shfl_xor(sq, off, 32);
  const float rs = rsqrtf(sq * (1.0f / 128.0f) + 1e-5f);
#pragma unroll
  for (int c = 0; c < 4; ++c) {
    const int ch   = h * D_HEAD + c * 32 + lane;
    const float y  = (v[c] - mu) * rs * wgt[ch] + bia[ch];
    rg[base + c * 32 + lane]  = y;
    rgb[base + c * 32 + lane] = (__bf16)y;
  }
}

__global__ void swish_mul_kernel(__bf16* __restrict__ hb, const float* __restrict__ g,
                                 const float* __restrict__ rg, int n) {
  const int i = blockIdx.x * blockDim.x + threadIdx.x;
  if (i < n) {
    const float gv = g[i];
    hb[i] = (__bf16)((gv / (1.0f + __expf(-gv))) * rg[i]);
  }
}

__global__ void cvt_bf16_kernel(__bf16* __restrict__ dst, const float* __restrict__ src,
                                int n) {
  const int i = blockIdx.x * blockDim.x + threadIdx.x;
  if (i < n) dst[i] = (__bf16)src[i];
}

// stT[h][e][d] = state[h][d][e]   (bf16)
__global__ void state_transpose_kernel(__bf16* __restrict__ stT,
                                       const float* __restrict__ state) {
  const int idx = blockIdx.x * 256 + threadIdx.x;  // h*16384 + e*128 + d
  const int d = idx & 127;
  const int e = (idx >> 7) & 127;
  const int h = idx >> 14;
  stT[idx] = (__bf16)state[((size_t)h * 128 + d) * 128 + e];
}

// ---------------------------------------------------------------------------
extern "C" void kernel_launch(void* const* d_in, const int* in_sizes, int n_in,
                              void* d_out, int out_size, void* d_ws, size_t ws_size,
                              hipStream_t stream) {
  (void)in_sizes; (void)n_in; (void)out_size; (void)ws_size;
  const float* x      = (const float*)d_in[0];
  const float* state  = (const float*)d_in[1];
  const float* W_qkv  = (const float*)d_in[2];
  const float* W_a    = (const float*)d_in[3];
  const float* b_a    = (const float*)d_in[4];
  const float* a_base = (const float*)d_in[5];
  const float* gn_w   = (const float*)d_in[6];
  const float* gn_b   = (const float*)d_in[7];
  const float* W_out  = (const float*)d_in[8];
  const float* W_gate = (const float*)d_in[9];
  const int*   offs   = (const int*)d_in[10];

  char* ws = (char*)d_ws;
  size_t o = 0;
  auto take = [&](size_t bytes) -> char* {
    char* p = ws + o;
    o = (o + bytes + 255) & ~(size_t)255;
    return p;
  };
  __bf16* x_bf    = (__bf16*)take((size_t)SQLEN * D_MODEL * 2);
  __bf16* Wqkv_bf = (__bf16*)take((size_t)QKV_N * D_MODEL * 2);
  __bf16* Wg_bf   = (__bf16*)take((size_t)D_MODEL * D_MODEL * 2);
  __bf16* Wo_bf   = (__bf16*)take((size_t)D_MODEL * D_MODEL * 2);
  __bf16* stT_bf  = (__bf16*)take((size_t)N_HEADS * D_HEAD * D_HEAD * 2);
  float*  qkv     = (float*) take((size_t)SQLEN * QKV_N * 4);
  float*  a_raw   = (float*) take((size_t)N_HEADS * SQLEN * 4);
  float*  ca      = (float*) take((size_t)N_HEADS * SQLEN * 4);
  __bf16* q_bf    = (__bf16*)take((size_t)N_HEADS * SQLEN * D_HEAD * 2);
  __bf16* k_bf    = (__bf16*)take((size_t)N_HEADS * SQLEN * D_HEAD * 2);
  __bf16* vT_bf   = (__bf16*)take((size_t)N_HEADS * D_HEAD * SQLEN * 2);
  float*  retb    = (float*) take((size_t)SQLEN * D_MODEL * 4);
  float*  ret_gn  = (float*) take((size_t)SQLEN * D_MODEL * 4);
  __bf16* rgn_bf  = (__bf16*)take((size_t)SQLEN * D_MODEL * 2);
  float*  G       = (float*) take((size_t)SQLEN * D_MODEL * 4);
  __bf16* H_bf    = (__bf16*)take((size_t)SQLEN * D_MODEL * 2);

  float* out    = (float*)d_out;
  float* nstate = out + (size_t)SQLEN * D_MODEL;

  // 1. precision conversions / transposes
  cvt_bf16_kernel<<<(SQLEN * D_MODEL) / 256, 256, 0, stream>>>(x_bf, x, SQLEN * D_MODEL);
  cvt_bf16_kernel<<<(QKV_N * D_MODEL) / 256, 256, 0, stream>>>(Wqkv_bf, W_qkv, QKV_N * D_MODEL);
  cvt_bf16_kernel<<<(D_MODEL * D_MODEL) / 256, 256, 0, stream>>>(Wg_bf, W_gate, D_MODEL * D_MODEL);
  cvt_bf16_kernel<<<(D_MODEL * D_MODEL) / 256, 256, 0, stream>>>(Wo_bf, W_out, D_MODEL * D_MODEL);
  state_transpose_kernel<<<(N_HEADS * D_HEAD * D_HEAD) / 256, 256, 0, stream>>>(stT_bf, state);

  // 2. qkv = x @ W_qkv^T  (WMMA, async-LDS pipelined)
  gemm_nt_wmma<<<dim3(QKV_N / 128, SQLEN / 64), 256, 0, stream>>>(
      qkv, x_bf, Wqkv_bf, SQLEN, QKV_N, D_MODEL);

  // 3. decays
  alpha_raw_kernel<<<(N_HEADS * SQLEN) / 8, 256, 0, stream>>>(a_raw, x, W_a, b_a, a_base);
  cumsum_kernel<<<N_HEADS, 32, 0, stream>>>(ca, a_raw);

  // 4. xpos / decay prep
  qkv_prep_kernel<<<(N_HEADS * SQLEN * 64) / 256, 256, 0, stream>>>(
      q_bf, k_bf, vT_bf, qkv, a_raw, offs);

  // 5. retention attention (WMMA)
  retention_attn_wmma<<<dim3(SQLEN / 16 / 8, N_HEADS), 256, 0, stream>>>(
      retb, q_bf, k_bf, vT_bf, stT_bf, ca);

  // 6. new state
  newstate_kernel<<<dim3(D_HEAD, N_HEADS), 128, 0, stream>>>(nstate, k_bf, qkv, ca, state);

  // 7. group norm
  groupnorm_kernel<<<(SQLEN * N_HEADS) / 8, 256, 0, stream>>>(ret_gn, rgn_bf, retb, gn_w, gn_b);

  // 8. gated output: out = (swish(ret_gn @ Wg^T) * ret_gn) @ Wo^T  (WMMA x2)
  gemm_nt_wmma<<<dim3(D_MODEL / 128, SQLEN / 64), 256, 0, stream>>>(
      G, rgn_bf, Wg_bf, SQLEN, D_MODEL, D_MODEL);
  swish_mul_kernel<<<(SQLEN * D_MODEL) / 256, 256, 0, stream>>>(
      H_bf, G, ret_gn, SQLEN * D_MODEL);
  gemm_nt_wmma<<<dim3(D_MODEL / 128, SQLEN / 64), 256, 0, stream>>>(
      out, H_bf, Wo_bf, SQLEN, D_MODEL, D_MODEL);
}
